// CrossAttention_48455821034193
// MI455X (gfx1250) — matrix operand order repaired and checked by reading, not compiled, not run on hardware
//
#include <hip/hip_runtime.h>
#include <hip/hip_bf16.h>

typedef __attribute__((ext_vector_type(16))) _Float16 v16h;
typedef __attribute__((ext_vector_type(8)))  _Float16 v8h;
typedef __attribute__((ext_vector_type(8)))  float    v8f;

#define B_    4
#define CQ_   128
#define CKV_  256
#define N_    4096
#define SCALE_ 0.25f   // (CQ/NUM_HEADS)^-0.5 = 16^-0.5

// ---------------- Q projection: Qh[b,n,o] = SCALE*(bq[o] + sum_c Wq[o,c] xq[b,c,n]) ----------------
__global__ void projq_kernel(const float* __restrict__ xq, const float* __restrict__ Wq,
                             const float* __restrict__ bq, _Float16* __restrict__ Qh) {
  int n = blockIdx.x * 256 + threadIdx.x;
  int o = blockIdx.y, b = blockIdx.z;
  const float* xp = xq + (size_t)b * CQ_ * N_ + n;
  const float* wp = Wq + (size_t)o * CQ_;
  float acc = bq[o];
#pragma unroll 8
  for (int c = 0; c < CQ_; ++c) acc += wp[c] * xp[(size_t)c * N_];
  Qh[((size_t)b * N_ + n) * CQ_ + o] = (_Float16)(acc * SCALE_);
}

// ---------------- K/V projection (fused; one pass over x_kv) ----------------
// Kh[b,n,o] (channels contiguous -> WMMA B of Q*K^T), Vh[b,o,n] (keys contiguous -> WMMA B of P*V)
__global__ void projkv_kernel(const float* __restrict__ xkv,
                              const float* __restrict__ Wk, const float* __restrict__ bk,
                              const float* __restrict__ Wv, const float* __restrict__ bv,
                              _Float16* __restrict__ Kh, _Float16* __restrict__ Vh) {
  int n = blockIdx.x * 256 + threadIdx.x;
  int o = blockIdx.y, b = blockIdx.z;
  const float* xp = xkv + (size_t)b * CKV_ * N_ + n;
  const float* wk = Wk + (size_t)o * CKV_;
  const float* wv = Wv + (size_t)o * CKV_;
  float ak = bk[o], av = bv[o];
#pragma unroll 8
  for (int c = 0; c < CKV_; ++c) {
    float x = xp[(size_t)c * N_];
    ak += wk[c] * x;
    av += wv[c] * x;
  }
  Kh[((size_t)b * N_ + n) * CQ_ + o] = (_Float16)ak;
  Vh[((size_t)b * CQ_ + o) * N_ + n] = (_Float16)av;
}

// async global -> LDS copy of 16 bytes per lane (CDNA5, ASYNCcnt-tracked)
__device__ __forceinline__ void async_ld16(uint32_t lds_off, const _Float16* gaddr) {
  asm volatile("global_load_async_to_lds_b128 %0, %1, off"
               :: "v"(lds_off), "v"(gaddr) : "memory");
}

// ---------------- Flash attention ----------------
// 8 waves/block, one 16-query block per wave; K/V tiles (64 keys) double-buffered in LDS
// and filled with async DMA shared by all 8 waves. Ah[b,c,n] = (softmax(Q^T K) V)[b,c,n].
__global__ void __launch_bounds__(256, 1)
flash_kernel(const _Float16* __restrict__ Qh, const _Float16* __restrict__ Kh,
             const _Float16* __restrict__ Vh, _Float16* __restrict__ Ah) {
  __shared__ __align__(16) _Float16 Kt[2][64][136];   // [key][chan], padded row
  __shared__ __align__(16) _Float16 Vt[2][128][72];   // [chan][key], padded row
  __shared__ __align__(16) _Float16 pb[8][16][72];    // per-wave P tile 16x64, padded

  const int wave = threadIdx.x >> 5;
  const int lane = threadIdx.x & 31;
  const int qb   = blockIdx.x * 8 + wave;   // global 16-row query block (b uniform per block)
  const int b    = qb >> 8;
  const int qbase = (qb & 255) << 4;
  const int nlo = lane & 15, hi = lane >> 4;

  const _Float16* Qb = Qh + (size_t)b * N_ * CQ_;
  const _Float16* Kb = Kh + (size_t)b * N_ * CQ_;
  const _Float16* Vb = Vh + (size_t)b * CQ_ * N_;

  // issue one 64-key K/V tile into LDS buffer `buf` (1024 * 16B units each, 4 per thread)
  auto stage = [&](int buf, int kb) {
#pragma unroll
    for (int it = 0; it < 4; ++it) {
      int unit = threadIdx.x + it * 256;
      int j = unit >> 4, c8 = (unit & 15) * 8;
      async_ld16((uint32_t)(size_t)&Kt[buf][j][c8], Kb + (size_t)(kb + j) * CQ_ + c8);
    }
#pragma unroll
    for (int it = 0; it < 4; ++it) {
      int unit = threadIdx.x + it * 256;
      int c = unit >> 3, k8 = (unit & 7) * 8;
      async_ld16((uint32_t)(size_t)&Vt[buf][c][k8], Vb + (size_t)c * N_ + kb + k8);
    }
  };

  // Q A-fragments (16x32 f16, interleaved K split per lane half)
  v16h qa[4];
#pragma unroll
  for (int cs = 0; cs < 4; ++cs) {
    const _Float16* p = Qb + (size_t)(qbase + nlo) * CQ_ + cs * 32 + hi * 8;
    union { v16h v; v8h h[2]; } u;
    u.h[0] = *(const v8h*)p;
    u.h[1] = *(const v8h*)(p + 16);
    qa[cs] = u.v;
  }

  v8f o[8];
  v8f mrow, lrow;
#pragma unroll
  for (int ct = 0; ct < 8; ++ct) o[ct] = (v8f){};
#pragma unroll
  for (int r = 0; r < 8; ++r) { mrow[r] = -3.0e38f; lrow[r] = 0.0f; }

  stage(0, 0);

  for (int kb = 0; kb < N_; kb += 64) {
    const int buf = (kb >> 6) & 1;
    asm volatile("s_wait_asynccnt 0" ::: "memory");   // our async stores to tile[buf] done
    __syncthreads();                                  // everyone's are done
    if (kb + 64 < N_) stage(buf ^ 1, kb + 64);        // overlap next DMA with compute

    // ---- S = (Q*SCALE) K^T : four 16x16 f32 tiles, K=128 in 4 WMMA steps each ----
    v8f s[4];
#pragma unroll
    for (int t = 0; t < 4; ++t) {
      v8f acc = (v8f){};
#pragma unroll
      for (int cs = 0; cs < 4; ++cs) {
        const _Float16* p = &Kt[buf][t * 16 + nlo][cs * 32 + hi * 8];
        union { v16h v; v8h h[2]; } u;
        u.h[0] = *(const v8h*)p;
        u.h[1] = *(const v8h*)(p + 16);
        acc = __builtin_amdgcn_wmma_f32_16x16x32_f16(false, qa[cs], false, u.v,
                                                     (short)0, acc, false, false);
      }
      s[t] = acc;
    }

    // ---- online softmax: combine all 4 tiles first, one reduction per chunk ----
    v8f cmax;
#pragma unroll
    for (int r = 0; r < 8; ++r)
      cmax[r] = fmaxf(fmaxf(s[0][r], s[1][r]), fmaxf(s[2][r], s[3][r]));
#pragma unroll
    for (int m = 1; m < 16; m <<= 1)
#pragma unroll
      for (int r = 0; r < 8; ++r) cmax[r] = fmaxf(cmax[r], __shfl_xor(cmax[r], m, 32));

    v8f mnew, alpha;
#pragma unroll
    for (int r = 0; r < 8; ++r) {
      mnew[r]  = fmaxf(mrow[r], cmax[r]);
      alpha[r] = __expf(mrow[r] - mnew[r]);
    }
    mrow = mnew;
#pragma unroll
    for (int r = 0; r < 8; ++r) lrow[r] *= alpha[r];
#pragma unroll
    for (int ct = 0; ct < 8; ++ct)
#pragma unroll
      for (int r = 0; r < 8; ++r) o[ct][r] *= alpha[r];

    // ---- P = exp(S - mnew); single combined row-sum reduction; stage P to LDS ----
    v8f psum = (v8f){};
#pragma unroll
    for (int t = 0; t < 4; ++t)
#pragma unroll
      for (int r = 0; r < 8; ++r) {
        s[t][r] = __expf(s[t][r] - mnew[r]);
        psum[r] += s[t][r];
      }
#pragma unroll
    for (int m = 1; m < 16; m <<= 1)
#pragma unroll
      for (int r = 0; r < 8; ++r) psum[r] += __shfl_xor(psum[r], m, 32);
#pragma unroll
    for (int r = 0; r < 8; ++r) lrow[r] += psum[r];

#pragma unroll
    for (int t = 0; t < 4; ++t)
#pragma unroll
      for (int r = 0; r < 8; ++r)
        pb[wave][hi * 8 + r][t * 16 + nlo] = (_Float16)s[t][r];
    asm volatile("s_wait_dscnt 0" ::: "memory");      // intra-wave P RAW

    // ---- O += P x V : two 16x32 P fragments x 8 channel tiles ----
#pragma unroll
    for (int u = 0; u < 2; ++u) {
      union { v16h v; v8h h[2]; } pu;
      pu.h[0] = *(const v8h*)&pb[wave][nlo][u * 32 + hi * 8];
      pu.h[1] = *(const v8h*)&pb[wave][nlo][u * 32 + 16 + hi * 8];
      v16h pa = pu.v;
#pragma unroll
      for (int ct = 0; ct < 8; ++ct) {
        const _Float16* vp = &Vt[buf][ct * 16 + nlo][u * 32 + hi * 8];
        union { v16h v; v8h h[2]; } vu;
        vu.h[0] = *(const v8h*)vp;
        vu.h[1] = *(const v8h*)(vp + 16);
        o[ct] = __builtin_amdgcn_wmma_f32_16x16x32_f16(false, pa, false, vu.v,
                                                       (short)0, o[ct], false, false);
      }
    }
  }

  // ---- normalize and store: Ah[b,c,n], 8 consecutive rows per lane -> one b128 store ----
  v8f inv;
#pragma unroll
  for (int r = 0; r < 8; ++r) inv[r] = 1.0f / lrow[r];
#pragma unroll
  for (int ct = 0; ct < 8; ++ct) {
    v8h oh;
#pragma unroll
    for (int r = 0; r < 8; ++r) oh[r] = (_Float16)(o[ct][r] * inv[r]);
    _Float16* op = Ah + ((size_t)b * CQ_ + ct * 16 + nlo) * N_ + qbase + hi * 8;
    *(v8h*)op = oh;
  }
}

// ---------------- Output projection + residual ----------------
__global__ void outproj_kernel(const _Float16* __restrict__ Ah, const float* __restrict__ Wo,
                               const float* __restrict__ bo, const float* __restrict__ xq,
                               float* __restrict__ out) {
  int n = blockIdx.x * 256 + threadIdx.x;
  int o = blockIdx.y, b = blockIdx.z;
  const float* wp = Wo + (size_t)o * CQ_;
  const _Float16* ap = Ah + (size_t)b * CQ_ * N_ + n;
  float acc = bo[o] + xq[((size_t)b * CQ_ + o) * N_ + n];
#pragma unroll 8
  for (int c = 0; c < CQ_; ++c) acc += wp[c] * (float)ap[(size_t)c * N_];
  out[((size_t)b * CQ_ + o) * N_ + n] = acc;
}

extern "C" void kernel_launch(void* const* d_in, const int* in_sizes, int n_in,
                              void* d_out, int out_size, void* d_ws, size_t ws_size,
                              hipStream_t stream) {
  const float* xq  = (const float*)d_in[0];
  const float* xkv = (const float*)d_in[1];
  const float* Wq  = (const float*)d_in[2];
  const float* bq  = (const float*)d_in[3];
  const float* Wk  = (const float*)d_in[4];
  const float* bk  = (const float*)d_in[5];
  const float* Wv  = (const float*)d_in[6];
  const float* bv  = (const float*)d_in[7];
  const float* Wo  = (const float*)d_in[8];
  const float* bo  = (const float*)d_in[9];
  float* out = (float*)d_out;

  // workspace: 4 x 4 MiB f16 buffers (Qh, Kh, Vh, Ah) = 16 MiB
  char* ws = (char*)d_ws;
  size_t seg = (size_t)B_ * N_ * CQ_ * sizeof(_Float16);
  _Float16* Qh = (_Float16*)(ws);
  _Float16* Kh = (_Float16*)(ws + seg);
  _Float16* Vh = (_Float16*)(ws + 2 * seg);
  _Float16* Ah = (_Float16*)(ws + 3 * seg);
  (void)in_sizes; (void)n_in; (void)out_size; (void)ws_size;

  dim3 blk(256);
  dim3 gproj(N_ / 256, CQ_, B_);
  projq_kernel  <<<gproj, blk, 0, stream>>>(xq, Wq, bq, Qh);
  projkv_kernel <<<gproj, blk, 0, stream>>>(xkv, Wk, bk, Wv, bv, Kh, Vh);
  flash_kernel  <<<dim3(B_ * (N_ / 16) / 8), blk, 0, stream>>>(Qh, Kh, Vh, Ah);
  outproj_kernel<<<gproj, blk, 0, stream>>>(Ah, Wo, bo, xq, out);
}
